// FlashAttentionVarlen_44143673868549
// MI455X (gfx1250) — matrix operand order repaired and checked by reading, not compiled, not run on hardware
//
#include <hip/hip_runtime.h>

typedef __attribute__((ext_vector_type(16))) _Float16 v16h;
typedef __attribute__((ext_vector_type(8)))  _Float16 v8h;
typedef __attribute__((ext_vector_type(2)))  __fp16   v2fp16;
typedef __attribute__((ext_vector_type(8)))  float    v8f;

#define NHEAD 16
#define HDIM  64
#define LCAP  256   // max seq len

// xor-16 lane swap as a single VALU op (v_permlanex16_b32, identity selects)
static __device__ __forceinline__ int swap16_i(int x) {
    return __builtin_amdgcn_permlanex16(x, x, 0x76543210, (int)0xFEDCBA98, false, false);
}
static __device__ __forceinline__ float swap16_f(float x) {
    union { float f; int i; } t; t.f = x;
    t.i = swap16_i(t.i);
    return t.f;
}
static __device__ __forceinline__ unsigned pack_f16x2(float a, float b) {
    union { v2fp16 h; unsigned u; } t;
    t.h = __builtin_amdgcn_cvt_pkrtz(a, b);   // v_cvt_pk_rtz_f16_f32
    return t.u;
}
static __device__ __forceinline__ v16h cat8(v8h lo, v8h hi) {
    return __builtin_shufflevector(lo, hi, 0,1,2,3,4,5,6,7,8,9,10,11,12,13,14,15);
}

__global__ __launch_bounds__(256)
void fa_varlen_kernel(const float* __restrict__ q,
                      const float* __restrict__ k,
                      const float* __restrict__ v,
                      const int*   __restrict__ cu,
                      float*       __restrict__ out)
{
    // 32 KB each: K as f16 [key][d], V^T as f16 [d][key]
    __shared__ _Float16 sK [LCAP * HDIM];
    __shared__ _Float16 sVt[HDIM * LCAP];

    const int bh  = blockIdx.x;
    const int b   = bh >> 4;            // / NHEAD
    const int h   = bh & (NHEAD - 1);
    const int tid = threadIdx.x;

    const int cu0 = cu[b];
    const int len = cu[b + 1] - cu0;
    if (len <= 0) return;                       // uniform per block
    const int lenPad = (len + 31) & ~31;        // pad to 32-key steps, zero-filled

    // ---------------- stage K (row-major) and V^T into LDS as f16 ----------------
    for (int idx = tid; idx < lenPad * (HDIM / 4); idx += 256) {
        const int key = idx >> 4;               // 16 float4-chunks per key
        const int d0  = (idx & 15) * 4;
        float4 kk = make_float4(0.f, 0.f, 0.f, 0.f);
        float4 vv = make_float4(0.f, 0.f, 0.f, 0.f);
        if (key < len) {
            const size_t base = ((size_t)(cu0 + key) * NHEAD + h) * HDIM + d0;
            kk = *(const float4*)(k + base);
            vv = *(const float4*)(v + base);
        }
        _Float16* kp = &sK[key * HDIM + d0];
        kp[0] = (_Float16)kk.x; kp[1] = (_Float16)kk.y;
        kp[2] = (_Float16)kk.z; kp[3] = (_Float16)kk.w;
        sVt[(d0 + 0) * LCAP + key] = (_Float16)vv.x;
        sVt[(d0 + 1) * LCAP + key] = (_Float16)vv.y;
        sVt[(d0 + 2) * LCAP + key] = (_Float16)vv.z;
        sVt[(d0 + 3) * LCAP + key] = (_Float16)vv.w;
    }
    __syncthreads();

    // ---------------- per-wave flash attention over 16-query tiles ----------------
    const int wave = tid >> 5;
    const int lane = tid & 31;
    const int qy   = lane & 15;     // row index inside 16-tile (query / key / d-row)
    const int hh   = lane >> 4;     // lane half
    // scores in log2-domain: fold 1/sqrt(64) * log2(e) into Q, use exp2
    const float qscale = 0.125f * 1.4426950408889634f;
    const int nqt = len >> 4;       // len is a multiple of 16 in this workload

    for (int qt = wave; qt < nqt; qt += 8) {
        // ---- load Q tile as B-operand of S^T = K * Q^T  (lane = query, elems = dims)
        v16h bq[2];
        {
            const int tok = cu0 + qt * 16 + qy;
            const float* qp = q + ((size_t)tok * NHEAD + h) * HDIM + hh * 16;
            #pragma unroll
            for (int c = 0; c < 2; ++c) {
                float tmp[16];
                *(float4*)(tmp +  0) = *(const float4*)(qp + c * 32 + 0);
                *(float4*)(tmp +  4) = *(const float4*)(qp + c * 32 + 4);
                *(float4*)(tmp +  8) = *(const float4*)(qp + c * 32 + 8);
                *(float4*)(tmp + 12) = *(const float4*)(qp + c * 32 + 12);
                v16h t;
                #pragma unroll
                for (int i = 0; i < 16; ++i) t[i] = (_Float16)(tmp[i] * qscale);
                bq[c] = t;
            }
        }

        v8f o[4] = {v8f{}, v8f{}, v8f{}, v8f{}};   // O^T accumulators (4 d-tiles)
        float m = -3.0e38f, lsum = 0.f;

        auto step = [&](int kb, bool domask) {
            // ---- S^T tiles (two 16-key tiles), K=64 dims in two 32-chunks
            v8f acc[2];
            #pragma unroll
            for (int t = 0; t < 2; ++t) {
                const _Float16* kr = &sK[(kb + t * 16 + qy) * HDIM];
                v8f s = v8f{};
                #pragma unroll
                for (int c = 0; c < 2; ++c) {
                    v8h lo = *(const v8h*)(kr + c * 32 + hh * 16);
                    v8h hi = *(const v8h*)(kr + c * 32 + hh * 16 + 8);
                    s = __builtin_amdgcn_wmma_f32_16x16x32_f16(
                            false, cat8(lo, hi), false, bq[c], (short)0, s, false, false);
                }
                acc[t] = s;
            }
            // ---- mask tail keys (lane/VGPR r holds key = kb + t*16 + 8*hh + r)
            if (domask) {
                #pragma unroll
                for (int t = 0; t < 2; ++t)
                    #pragma unroll
                    for (int r = 0; r < 8; ++r)
                        if (kb + t * 16 + hh * 8 + r >= len) acc[t][r] = -1.0e9f;
            }
            // ---- online softmax (row = query lives in lanes q and q+16)
            float mloc = acc[0][0];
            #pragma unroll
            for (int r = 1; r < 8; ++r) mloc = fmaxf(mloc, acc[0][r]);
            #pragma unroll
            for (int r = 0; r < 8; ++r) mloc = fmaxf(mloc, acc[1][r]);
            mloc = fmaxf(mloc, swap16_f(mloc));
            const float mnew  = fmaxf(m, mloc);
            const float alpha = __builtin_amdgcn_exp2f(m - mnew);
            m = mnew;
            float p0[8], p1[8], ls = 0.f;
            #pragma unroll
            for (int r = 0; r < 8; ++r) { p0[r] = __builtin_amdgcn_exp2f(acc[0][r] - mnew); ls += p0[r]; }
            #pragma unroll
            for (int r = 0; r < 8; ++r) { p1[r] = __builtin_amdgcn_exp2f(acc[1][r] - mnew); ls += p1[r]; }
            ls += swap16_f(ls);
            lsum = lsum * alpha + ls;
            #pragma unroll
            for (int vt = 0; vt < 4; ++vt)
                #pragma unroll
                for (int e = 0; e < 8; ++e) o[vt][e] *= alpha;

            // ---- P^T as B-operand: cross-half swap only (lane=query, VGPR j = keys 16*hh+2j,2j+1)
            union { unsigned u[8]; v16h vv; } bp;
            #pragma unroll
            for (int i = 0; i < 4; ++i) {
                unsigned P0 = pack_f16x2(p0[2 * i], p0[2 * i + 1]);
                unsigned P1 = pack_f16x2(p1[2 * i], p1[2 * i + 1]);
                unsigned s0 = (unsigned)swap16_i((int)P0);
                unsigned s1 = (unsigned)swap16_i((int)P1);
                bp.u[i]     = hh ? s1 : P0;   // keys 16*hh + 0..7  from lane (q,0)
                bp.u[i + 4] = hh ? P1 : s0;   // keys 16*hh + 8..15 from lane (q,1)
            }
            // ---- O^T += V^T * P^T  (A = V^T tile: row = d, elems = keys)
            #pragma unroll
            for (int vt = 0; vt < 4; ++vt) {
                const _Float16* vr = &sVt[(vt * 16 + qy) * LCAP + kb];
                v8h lo = *(const v8h*)(vr + hh * 16);
                v8h hi = *(const v8h*)(vr + hh * 16 + 8);
                o[vt] = __builtin_amdgcn_wmma_f32_16x16x32_f16(
                            false, cat8(lo, hi), false, bp.vv, (short)0, o[vt], false, false);
            }
        };

        const int lenFull = len & ~31;
        for (int kb = 0; kb < lenFull; kb += 32) step(kb, false);
        if (lenFull < len) step(lenFull, true);   // masked tail (never taken for len%32==0)

        // ---- epilogue: O^T lane (q,hh), VGPR r -> d = vt*16 + 8*hh + r
        const int row = qt * 16 + qy;
        if (row < len) {
            const float inv = 1.0f / lsum;
            float* op = out + ((size_t)(cu0 + row) * NHEAD + h) * HDIM;
            #pragma unroll
            for (int vt = 0; vt < 4; ++vt) {
                const int d0 = vt * 16 + hh * 8;
                *(float4*)(op + d0) =
                    make_float4(o[vt][0] * inv, o[vt][1] * inv, o[vt][2] * inv, o[vt][3] * inv);
                *(float4*)(op + d0 + 4) =
                    make_float4(o[vt][4] * inv, o[vt][5] * inv, o[vt][6] * inv, o[vt][7] * inv);
            }
        }
    }
}

extern "C" void kernel_launch(void* const* d_in, const int* in_sizes, int n_in,
                              void* d_out, int out_size, void* d_ws, size_t ws_size,
                              hipStream_t stream) {
    (void)n_in; (void)out_size; (void)d_ws; (void)ws_size;
    const float* q  = (const float*)d_in[0];
    const float* k  = (const float*)d_in[1];
    const float* v  = (const float*)d_in[2];
    const int*   cu = (const int*)d_in[3];
    const int B = in_sizes[3] - 1;           // cu_seqlens has B+1 entries
    fa_varlen_kernel<<<dim3(B * NHEAD), dim3(256), 0, stream>>>(q, k, v, cu, (float*)d_out);
}
